// GraphSAGE_34342558499454
// MI455X (gfx1250) — matrix operand order repaired and checked by reading, not compiled, not run on hardware
//
#include <hip/hip_runtime.h>
#include <hip/hip_bf16.h>

#define N_NODES 100000
#define IN_C 32
#define HID_C 64
#define OUT_C 32

typedef __attribute__((ext_vector_type(16))) __bf16 v16bf;
typedef __attribute__((ext_vector_type(8)))  float  v8f;

// ---------------------------------------------------------------- init: zero deg + agg buffers
__global__ void sage_zero(float* __restrict__ p, long long n) {
    long long i = (long long)blockIdx.x * blockDim.x + threadIdx.x;
    if (i < n) p[i] = 0.0f;
}

// ---------------------------------------------------------------- pre-pack weights into WMMA B-fragment layout
// B (32x16 bf16) layout: lane = col N (mod 16); lanes 0-15 hold K=0..7,16..23,
// lanes 16-31 hold K=8..15,24..31. Fragment stored lane-major: pw[(frag*32 + lane)*16 + e].
// layer1: frag f = kc*4 + nt (kc: 0=W1l, 1=W1r; nt: 4 col tiles of 64 outputs), K=32 each
// layer2: frag f = kc*2 + nt (kc 0,1 = W2l K-halves; kc 2,3 = W2r K-halves), K=64 per W
__global__ void sage_pack_w(const float* __restrict__ W1l, const float* __restrict__ W1r,
                            const float* __restrict__ W2l, const float* __restrict__ W2r,
                            __bf16* __restrict__ pw1, __bf16* __restrict__ pw2) {
    int t = blockIdx.x * 256 + threadIdx.x;          // 0..8191
    int e = t & 15;
    int L = (t >> 4) & 31;
    int f = (t >> 9) & 7;
    int khalf = (L >> 4) << 3;                       // +8 for upper lanes
    int ncol = L & 15;
    if (t < 4096) {                                  // layer 1 fragments
        int kc = f >> 2, nt = f & 3;
        int n = nt * 16 + ncol;
        int k = ((e < 8) ? e : e + 8) + khalf;
        const float* W = kc ? W1r : W1l;
        pw1[t] = (__bf16)W[n * IN_C + k];
    } else {                                         // layer 2 fragments
        int kc = f >> 1, nt = f & 1;
        int n = nt * 16 + ncol;
        int k = (kc & 1) * 32 + ((e < 8) ? e : e + 8) + khalf;
        const float* W = (kc < 2) ? W2l : W2r;
        pw2[t - 4096] = (__bf16)W[n * HID_C + k];
    }
}

// ---------------------------------------------------------------- scatter layer 1 (32 channels == wave32)
__global__ void sage_scatter1(const int* __restrict__ src, const int* __restrict__ dst,
                              const float* __restrict__ x, float* __restrict__ agg1,
                              float* __restrict__ deg, int nE) {
    int e = blockIdx.x * 8 + (threadIdx.x >> 5);
    if (e >= nE) return;
    int c = threadIdx.x & 31;
    int s = src[e], d = dst[e];
    __hip_atomic_fetch_add(&agg1[(long long)d * IN_C + c], x[(long long)s * IN_C + c],
                           __ATOMIC_RELAXED, __HIP_MEMORY_SCOPE_AGENT);
    if (c == 0)
        __hip_atomic_fetch_add(&deg[d], 1.0f, __ATOMIC_RELAXED, __HIP_MEMORY_SCOPE_AGENT);
}

// ---------------------------------------------------------------- scatter layer 2 (64 channels)
__global__ void sage_scatter2(const int* __restrict__ src, const int* __restrict__ dst,
                              const float* __restrict__ h, float* __restrict__ agg2, int nE) {
    int e = blockIdx.x * 4 + (threadIdx.x >> 6);
    if (e >= nE) return;
    int c = threadIdx.x & 63;
    int s = src[e], d = dst[e];
    __hip_atomic_fetch_add(&agg2[(long long)d * HID_C + c], h[(long long)s * HID_C + c],
                           __ATOMIC_RELAXED, __HIP_MEMORY_SCOPE_AGENT);
}

// ---------------------------------------------------------------- layer 1: h = relu([mean|x] @ [W1l;W1r]^T + b1)
__global__ __launch_bounds__(128) void sage_layer1(
    const float* __restrict__ x,   const float* __restrict__ agg1,
    const float* __restrict__ deg, const v16bf* __restrict__ pw1,
    const float* __restrict__ b1,  float* __restrict__ h) {
    const int wave = threadIdx.x >> 5;
    const int lane = threadIdx.x & 31;
    const int tile = blockIdx.x * 4 + wave;
    if (tile >= N_NODES / 16) return;                 // wave-uniform: EXEC stays full
    const int m0    = tile * 16;
    const int mrow  = m0 + (lane & 15);               // A-matrix row for this lane
    const int ahalf = (lane >> 4) << 3;               // +8 K-offset for lanes 16-31 (A)
    const int ncol  = lane & 15;
    const float invd = 1.0f / fmaxf(deg[mrow], 1.0f);

    v8f acc[4] = {v8f{}, v8f{}, v8f{}, v8f{}};

    // K-chunk 0: mean-aggregated neighbors (scaled agg) through packed W1_l
    {
        const float* rowA = agg1 + (long long)mrow * IN_C;
        v16bf a;
        #pragma unroll
        for (int e = 0; e < 16; ++e) {
            int k = ((e < 8) ? e : e + 8) + ahalf;    // 16-bit A layout (16x32)
            a[e] = (__bf16)(rowA[k] * invd);
        }
        #pragma unroll
        for (int nt = 0; nt < 4; ++nt) {
            v16bf b = pw1[nt * 32 + lane];            // frag kc=0
            acc[nt] = __builtin_amdgcn_wmma_f32_16x16x32_bf16(
                false, a, false, b, (short)0, acc[nt], false, false);
        }
    }
    // K-chunk 1: root features through packed W1_r
    {
        const float* rowA = x + (long long)mrow * IN_C;
        v16bf a;
        #pragma unroll
        for (int e = 0; e < 16; ++e) {
            int k = ((e < 8) ? e : e + 8) + ahalf;
            a[e] = (__bf16)rowA[k];
        }
        #pragma unroll
        for (int nt = 0; nt < 4; ++nt) {
            v16bf b = pw1[(4 + nt) * 32 + lane];      // frag kc=1
            acc[nt] = __builtin_amdgcn_wmma_f32_16x16x32_bf16(
                false, a, false, b, (short)0, acc[nt], false, false);
        }
    }
    // D layout: VGPR r -> M = r (+8 for upper lanes), N = lane&15; bias + relu
    const int mbase = (lane >> 4) << 3;
    #pragma unroll
    for (int nt = 0; nt < 4; ++nt) {
        int n = nt * 16 + ncol;
        float bias = b1[n];
        #pragma unroll
        for (int r = 0; r < 8; ++r) {
            int m = m0 + mbase + r;
            h[(long long)m * HID_C + n] = fmaxf(acc[nt][r] + bias, 0.0f);
        }
    }
}

// ---------------------------------------------------------------- layer 2: out = [mean2|h] @ [W2l;W2r]^T + b2
__global__ __launch_bounds__(128) void sage_layer2(
    const float* __restrict__ h,   const float* __restrict__ agg2,
    const float* __restrict__ deg, const v16bf* __restrict__ pw2,
    const float* __restrict__ b2,  float* __restrict__ out) {
    const int wave = threadIdx.x >> 5;
    const int lane = threadIdx.x & 31;
    const int tile = blockIdx.x * 4 + wave;
    if (tile >= N_NODES / 16) return;
    const int m0    = tile * 16;
    const int mrow  = m0 + (lane & 15);
    const int ahalf = (lane >> 4) << 3;
    const int ncol  = lane & 15;
    const float invd = 1.0f / fmaxf(deg[mrow], 1.0f);

    v8f acc[2] = {v8f{}, v8f{}};

    #pragma unroll
    for (int kc = 0; kc < 4; ++kc) {
        const bool meanPart = (kc < 2);
        const int kbase = (kc & 1) * 32;              // offset within the 64-wide operand
        const float* rowA = (meanPart ? agg2 : h) + (long long)mrow * HID_C + kbase;
        const float scale = meanPart ? invd : 1.0f;
        v16bf a;
        #pragma unroll
        for (int e = 0; e < 16; ++e) {
            int k = ((e < 8) ? e : e + 8) + ahalf;
            a[e] = (__bf16)(rowA[k] * scale);
        }
        #pragma unroll
        for (int nt = 0; nt < 2; ++nt) {
            v16bf b = pw2[(kc * 2 + nt) * 32 + lane];
            acc[nt] = __builtin_amdgcn_wmma_f32_16x16x32_bf16(
                false, a, false, b, (short)0, acc[nt], false, false);
        }
    }

    const int mbase = (lane >> 4) << 3;
    #pragma unroll
    for (int nt = 0; nt < 2; ++nt) {
        int n = nt * 16 + ncol;
        float bias = b2[n];
        #pragma unroll
        for (int r = 0; r < 8; ++r) {
            int m = m0 + mbase + r;
            out[(long long)m * OUT_C + n] = acc[nt][r] + bias;
        }
    }
}

// ----------------------------------------------------------------
extern "C" void kernel_launch(void* const* d_in, const int* in_sizes, int n_in,
                              void* d_out, int out_size, void* d_ws, size_t ws_size,
                              hipStream_t stream) {
    const float* x   = (const float*)d_in[0];
    const float* W1l = (const float*)d_in[1];
    const float* W1r = (const float*)d_in[2];
    const float* b1  = (const float*)d_in[3];
    const float* W2l = (const float*)d_in[4];
    const float* W2r = (const float*)d_in[5];
    const float* b2  = (const float*)d_in[6];
    const int*   ei  = (const int*)d_in[7];           // [2, E] row-major
    const int E = in_sizes[7] / 2;
    const int* src = ei;
    const int* dst = ei + E;

    float* ws   = (float*)d_ws;
    float* deg  = ws;                                          // N floats
    float* agg1 = deg  + N_NODES;                              // N*32
    float* agg2 = agg1 + (long long)N_NODES * IN_C;            // N*64
    float* h    = agg2 + (long long)N_NODES * HID_C;           // N*64
    __bf16* pw1 = (__bf16*)(h + (long long)N_NODES * HID_C);   // 4096 bf16 (8 KB)
    __bf16* pw2 = pw1 + 4096;                                  // 4096 bf16 (8 KB)
    float* outp = (float*)d_out;

    // zero deg + agg1 + agg2 (h and out are fully overwritten)
    long long zcnt = (long long)N_NODES * (1 + IN_C + HID_C);
    sage_zero<<<(unsigned)((zcnt + 255) / 256), 256, 0, stream>>>(deg, zcnt);

    // pre-pack weights into WMMA B-fragment layout (16 KB total, L2-resident)
    sage_pack_w<<<32, 256, 0, stream>>>(W1l, W1r, W2l, W2r, pw1, pw2);

    // layer 1 scatter: 8 edges / 256-thread block, lane == channel
    sage_scatter1<<<(E + 7) / 8, 256, 0, stream>>>(src, dst, x, agg1, deg, E);

    // layer 1 dense: 6250 tiles of 16 nodes, 4 waves / block
    const int nTiles = N_NODES / 16;
    sage_layer1<<<(nTiles + 3) / 4, 128, 0, stream>>>(x, agg1, deg, (const v16bf*)pw1, b1, h);

    // layer 2 scatter: 4 edges / 256-thread block, 64 channels
    sage_scatter2<<<(E + 3) / 4, 256, 0, stream>>>(src, dst, h, agg2, E);

    // layer 2 dense
    sage_layer2<<<(nTiles + 3) / 4, 128, 0, stream>>>(h, agg2, deg, (const v16bf*)pw2, b2, outp);
}
